// R2MoE_3221225472408
// MI455X (gfx1250) — matrix operand order repaired and checked by reading, not compiled, not run on hardware
//
#include <hip/hip_runtime.h>
#include <hip/hip_bf16.h>

// ---------------- problem constants (from setup_inputs) ----------------
constexpr int BATCH = 8;
constexpr int SEQ   = 8192;
constexpr int DIN   = 768;
constexpr int DOUT  = 768;
constexpr int POOL  = 5;
constexpr int RANK  = 8;
constexpr int TOPK  = 3;
constexpr int ROWS  = BATCH * SEQ;          // 65536 tokens
constexpr float SCALE = 1.0f;

typedef __attribute__((ext_vector_type(16))) __bf16 v16bf;
typedef __attribute__((ext_vector_type(8)))  float  v8f;
typedef __attribute__((ext_vector_type(2)))  int    v2i;

union BF16Frag { v16bf v; uint4 q[2]; };    // 32 bytes = 8 VGPRs

#if __has_builtin(__builtin_amdgcn_global_load_async_to_lds_b64)
#define HAVE_ASYNC_LDS 1
#define AS1 __attribute__((address_space(1)))
#define AS3 __attribute__((address_space(3)))
#endif

// RNE f32 -> bf16 bits (no NaN/Inf inputs in this problem)
__device__ __forceinline__ unsigned bf16_rne(float f) {
    unsigned u = __float_as_uint(f);
    u += 0x7FFFu + ((u >> 16) & 1u);
    return u >> 16;
}
// split f = hi + lo with hi = RNE bf16(f), lo = RNE bf16(f - hi)
__device__ __forceinline__ void split_bf16(float f, unsigned& hb, unsigned& lb) {
    hb = bf16_rne(f);
    float hf = __uint_as_float(hb << 16);
    lb = bf16_rne(f - hf);
}

// ---------------------------------------------------------------------
// K1: deterministic column-sum partials (block b -> rows [b*256, b*256+256))
// ---------------------------------------------------------------------
__global__ __launch_bounds__(256) void k_reduce_partial(
    const float* __restrict__ x, float* __restrict__ partials)
{
    const int b = blockIdx.x;
    const int t = threadIdx.x;
    const float* base = x + (size_t)b * 256 * DIN;
    float a0 = 0.f, a1 = 0.f, a2 = 0.f;
    for (int r = 0; r < 256; ++r) {
        const float* rp = base + (size_t)r * DIN;
        a0 += rp[t];
        a1 += rp[t + 256];
        a2 += rp[t + 512];
    }
    float* pr = partials + (size_t)b * DIN;
    pr[t]       = a0;
    pr[t + 256] = a1;
    pr[t + 512] = a2;
}

// K1b: fixed-order fold -> xbar (includes 1/(B*S))
__global__ __launch_bounds__(256) void k_reduce_final(
    const float* __restrict__ partials, float* __restrict__ xbar)
{
    const int col = blockIdx.x * 256 + threadIdx.x;
    if (col >= DIN) return;
    float s = 0.f;
    for (int b = 0; b < 256; ++b) s += partials[(size_t)b * DIN + col];
    xbar[col] = s * (1.0f / (float)ROWS);
}

// ---------------------------------------------------------------------
// K2: one wave. omegas -> top-k -> softmax gate. Writes meta_i / meta_g.
// ---------------------------------------------------------------------
__global__ __launch_bounds__(32) void k_gate(
    const float* __restrict__ xbar, const float* __restrict__ route,
    const int* __restrict__ tid_p, int* __restrict__ meta_i,
    float* __restrict__ meta_g)
{
    __shared__ float red[32];
    __shared__ float omegas[POOL];
    const int lane = threadIdx.x;
    const int task_id = tid_p[0];

    for (int p = 0; p < POOL; ++p) {
        float part = 0.f;
        for (int i = lane; i < DIN; i += 32) {
            float rs = 0.f;
            for (int tt = 0; tt < task_id; ++tt)
                rs += route[(size_t)tt * DIN * POOL + (size_t)i * POOL + p];
            part += xbar[i] * rs;
        }
        red[lane] = part;
        __syncthreads();
        if (lane == 0) {
            float s = 0.f;
            for (int j = 0; j < 32; ++j) s += red[j];
            omegas[p] = s;
        }
        __syncthreads();
    }

    if (lane == 0) {
        int k = task_id < TOPK ? task_id : TOPK;
        int navail = (task_id < POOL - 1) ? task_id : (POOL - 1);
        if (k > navail) k = navail;

        bool used[POOL];
        for (int q = 0; q < POOL; ++q) used[q] = false;
        float G[TOPK + 1];
        int   I[TOPK + 1];
        G[0] = omegas[0];
        I[0] = 0;
        for (int j = 1; j <= k; ++j) {   // lax.top_k: descending, low idx on tie
            int best = -1; float bv = -3.4e38f;
            for (int q = 1; q < POOL && q <= task_id; ++q)
                if (!used[q] && omegas[q] > bv) { bv = omegas[q]; best = q; }
            used[best] = true;
            G[j] = bv; I[j] = best;
        }
        float mx = G[0];
        for (int j = 1; j <= k; ++j) mx = fmaxf(mx, G[j]);
        float den = 0.f;
        for (int j = 0; j <= k; ++j) den += expf(G[j] - mx);
        meta_i[0] = k + 1;
        for (int j = 0; j <= k; ++j) {
            meta_i[1 + j] = I[j];
            meta_g[j] = expf(G[j] - mx) / den;
        }
        for (int j = k + 1; j <= TOPK; ++j) { meta_i[1 + j] = 0; meta_g[j] = 0.f; }
    }
}

// ---------------------------------------------------------------------
// K3: effective weights, pre-transposed to [o][i] and pre-split to bf16 hi/lo.
//   sel0[o][i] = W[o][i]
//   sel1[o][i] = W[o][i] + SCALE * sum_e gate[e] * (down[Ie] @ up[Ie])[i][o]
// ---------------------------------------------------------------------
__global__ __launch_bounds__(256) void k_build_weff(
    const float* __restrict__ W, const float* __restrict__ down,
    const float* __restrict__ up, const int* __restrict__ meta_i,
    const float* __restrict__ meta_g,
    unsigned short* __restrict__ weff_hi, unsigned short* __restrict__ weff_lo)
{
    const int idx = blockIdx.x * 256 + threadIdx.x;
    if (idx >= DIN * DOUT) return;
    const int o = idx / DIN;
    const int i = idx % DIN;
    const float wt = W[(size_t)o * DIN + i];

    const int ne = meta_i[0];
    float m = 0.f;
    for (int e = 0; e < ne; ++e) {
        const int Ie = meta_i[1 + e];
        const float* dn = down + (size_t)Ie * DIN * RANK + (size_t)i * RANK;
        const float* uq = up   + (size_t)Ie * RANK * DOUT + o;
        float s = 0.f;
        #pragma unroll
        for (int r = 0; r < RANK; ++r) s += dn[r] * uq[(size_t)r * DOUT];
        m += meta_g[e] * s;
    }
    unsigned hb, lb;
    split_bf16(wt, hb, lb);
    weff_hi[idx] = (unsigned short)hb;
    weff_lo[idx] = (unsigned short)lb;
    split_bf16(wt + SCALE * m, hb, lb);
    weff_hi[DIN * DOUT + idx] = (unsigned short)hb;
    weff_lo[DIN * DOUT + idx] = (unsigned short)lb;
}

// ---------------------------------------------------------------------
// K4: main GEMM, error-compensated split-bf16 (bf16x3) via
// V_WMMA_F32_16X16X32_BF16:  x @ Weff[sel]  ~=  hh + hl + lh  (lo*lo dropped)
//
// Block tile: 256 rows x 64 cols, K panel 32, 8 waves. Wave = 32x64 strip
// (2 row-tiles x 4 col-tiles of 16x16 => acc[2][4]); 24 WMMA per 24
// ds_load_b128 per panel -> LDS bandwidth balanced against the XDL pipe.
// B panels stream global->LDS via GLOBAL_LOAD_ASYNC_TO_LDS_B64 (ASYNCcnt).
//
// Fragment layouts (CDNA5 ISA 7.12.2, 16-bit):
//   A (16x32): lane l: M=l%16, h=l/16; VGPR0-3 = K 8h..8h+7, VGPR4-7 = K 16+8h..+7
//       -> LDS row-major [row][k], two b128 chunks at k = 8h and 16+8h
//   B (32x16): lane l: N=l%16, h=l/16; VGPR0-3 = K 8h..8h+7, VGPR4-7 = K 16+8h..+7
//       -> LDS col-major [col][k], two b128 chunks at k = 8h, 16+8h
//   D: lane l, vgpr v -> M = v + 8*(l/16), N = l%16
// ---------------------------------------------------------------------
#define LDAK 40     // padded K stride (bf16 elems) for both A rows and B cols

__global__ __launch_bounds__(256) void k_gemm(
    const float* __restrict__ x,
    const unsigned short* __restrict__ weff_hi,
    const unsigned short* __restrict__ weff_lo,
    float* __restrict__ out)
{
    __shared__ alignas(16) unsigned short lds_ah[256 * LDAK];   // 20.0 KB
    __shared__ alignas(16) unsigned short lds_al[256 * LDAK];   // 20.0 KB
    __shared__ alignas(16) unsigned short lds_bh[64 * LDAK];    //  5.0 KB
    __shared__ alignas(16) unsigned short lds_bl[64 * LDAK];    //  5.0 KB

    const int t    = threadIdx.x;
    const int wave = t >> 5;
    const int lane = t & 31;
    const int h    = lane >> 4;               // lane half (K split)
    const int n    = lane & 15;               // M for A frags, N for B/D frags

    const int ob       = blockIdx.x * 64;
    const int row_base = blockIdx.y * 256;
    const int sel      = (row_base >= ROWS / 2) ? 1 : 0;   // batches 0..3: no delta
    const unsigned short* whi = weff_hi + (size_t)sel * DIN * DOUT;
    const unsigned short* wlo = weff_lo + (size_t)sel * DIN * DOUT;

    v8f acc[2][4] = {};

    for (int kb = 0; kb < DIN; kb += 32) {
        // ---- stage B: Weff_hi/lo[ob..+64][kb..+32] -> LDS [col][k] ----
        #pragma unroll
        for (int i = 0; i < 2; ++i) {
            const int idx = t + i * 256;
            const int col = idx >> 3;                 // 0..63 (o-local)
            const int c4  = (idx & 7) << 2;           // k offset, 4 bf16 chunks
            const size_t g = (size_t)(ob + col) * DIN + kb + c4;
#ifdef HAVE_ASYNC_LDS
            // ASYNCcnt-tracked direct global->LDS DMA (bypasses VGPRs)
            __builtin_amdgcn_global_load_async_to_lds_b64(
                (AS1 v2i*)(whi + g), (AS3 v2i*)&lds_bh[col * LDAK + c4], 0, 0);
            __builtin_amdgcn_global_load_async_to_lds_b64(
                (AS1 v2i*)(wlo + g), (AS3 v2i*)&lds_bl[col * LDAK + c4], 0, 0);
#else
            *(uint2*)&lds_bh[col * LDAK + c4] = *(const uint2*)(whi + g);
            *(uint2*)&lds_bl[col * LDAK + c4] = *(const uint2*)(wlo + g);
#endif
        }
        // ---- stage A: x[row_base..+256][kb..+32], split to bf16 hi/lo ----
        #pragma unroll
        for (int i = 0; i < 8; ++i) {
            const int idx = t + i * 256;
            const int row = idx >> 3;
            const int c4  = (idx & 7) << 2;
            const float4 v = *(const float4*)(x + (size_t)(row_base + row) * DIN + kb + c4);
            unsigned h0, l0, h1, l1, h2, l2, h3, l3;
            split_bf16(v.x, h0, l0);
            split_bf16(v.y, h1, l1);
            split_bf16(v.z, h2, l2);
            split_bf16(v.w, h3, l3);
            *(uint2*)&lds_ah[row * LDAK + c4] = make_uint2(h0 | (h1 << 16), h2 | (h3 << 16));
            *(uint2*)&lds_al[row * LDAK + c4] = make_uint2(l0 | (l1 << 16), l2 | (l3 << 16));
        }
        // prefetch next A panel (global_prefetch_b8)
        if (kb + 32 < DIN) {
            const int row = t >> 3;
            const int c4  = (t & 7) << 2;
            __builtin_prefetch(x + (size_t)(row_base + row) * DIN + kb + 32 + c4, 0, 0);
        }
#ifdef HAVE_ASYNC_LDS
#if __has_builtin(__builtin_amdgcn_s_wait_asynccnt)
        __builtin_amdgcn_s_wait_asynccnt(0);
#else
        asm volatile("s_wait_asynccnt 0" ::: "memory");
#endif
#endif
        __syncthreads();

        // ---- fragments ----
        BF16Frag ah[2], al[2];
        #pragma unroll
        for (int mt = 0; mt < 2; ++mt) {
            const int arow = wave * 32 + mt * 16 + n;
            ah[mt].q[0] = *(const uint4*)&lds_ah[arow * LDAK + 8 * h];
            ah[mt].q[1] = *(const uint4*)&lds_ah[arow * LDAK + 16 + 8 * h];
            al[mt].q[0] = *(const uint4*)&lds_al[arow * LDAK + 8 * h];
            al[mt].q[1] = *(const uint4*)&lds_al[arow * LDAK + 16 + 8 * h];
        }
        BF16Frag bh[4], bl[4];
        #pragma unroll
        for (int nt = 0; nt < 4; ++nt) {
            const int base = (nt * 16 + n) * LDAK + 8 * h;
            bh[nt].q[0] = *(const uint4*)&lds_bh[base];
            bh[nt].q[1] = *(const uint4*)&lds_bh[base + 16];
            bl[nt].q[0] = *(const uint4*)&lds_bl[base];
            bl[nt].q[1] = *(const uint4*)&lds_bl[base + 16];
        }

        // ---- compute: combo-major; each acc has 8 independent WMMAs between reuses
        #pragma unroll
        for (int mt = 0; mt < 2; ++mt)
            #pragma unroll
            for (int nt = 0; nt < 4; ++nt)
                acc[mt][nt] = __builtin_amdgcn_wmma_f32_16x16x32_bf16(
                    false, ah[mt].v, false, bh[nt].v, (short)0, acc[mt][nt], false, false);
        #pragma unroll
        for (int mt = 0; mt < 2; ++mt)
            #pragma unroll
            for (int nt = 0; nt < 4; ++nt)
                acc[mt][nt] = __builtin_amdgcn_wmma_f32_16x16x32_bf16(
                    false, ah[mt].v, false, bl[nt].v, (short)0, acc[mt][nt], false, false);
        #pragma unroll
        for (int mt = 0; mt < 2; ++mt)
            #pragma unroll
            for (int nt = 0; nt < 4; ++nt)
                acc[mt][nt] = __builtin_amdgcn_wmma_f32_16x16x32_bf16(
                    false, al[mt].v, false, bh[nt].v, (short)0, acc[mt][nt], false, false);

        __syncthreads();
    }

    // ---- epilogue: D layout M = v + 8*h, N = n ----
    #pragma unroll
    for (int mt = 0; mt < 2; ++mt) {
        #pragma unroll
        for (int nt = 0; nt < 4; ++nt) {
            const int col = ob + nt * 16 + n;
            #pragma unroll
            for (int v = 0; v < 8; ++v) {
                const int row = row_base + wave * 32 + mt * 16 + (h << 3) + v;
                out[(size_t)row * DOUT + col] = acc[mt][nt][v];
            }
        }
    }
}

// ---------------------------------------------------------------------
extern "C" void kernel_launch(void* const* d_in, const int* in_sizes, int n_in,
                              void* d_out, int out_size, void* d_ws, size_t ws_size,
                              hipStream_t stream)
{
    const float* x          = (const float*)d_in[0];   // [8, 8192, 768]
    const float* W          = (const float*)d_in[1];   // [768, 768]
    const float* lora_down  = (const float*)d_in[2];   // [5, 768, 8]
    const float* lora_up    = (const float*)d_in[3];   // [5, 8, 768]
    const float* lora_route = (const float*)d_in[4];   // [5, 768, 5]
    const int*   task_id    = (const int*)d_in[5];     // scalar
    float*       out        = (float*)d_out;           // [65536, 768]

    // ws layout: xbar(768 f) | meta_i(8 int) meta_g(8 f) | partials(256*768 f)
    //            | weff_hi(2*768*768 bf16) | weff_lo(2*768*768 bf16)   (~5.5 MB)
    float* wsf      = (float*)d_ws;
    float* xbar     = wsf;
    int*   meta_i   = (int*)(wsf + DIN);
    float* meta_g   = wsf + DIN + 8;
    float* partials = wsf + 1024;
    unsigned short* weff_hi = (unsigned short*)(wsf + 1024 + 256 * DIN);
    unsigned short* weff_lo = weff_hi + 2 * DIN * DOUT;

    k_reduce_partial<<<256, 256, 0, stream>>>(x, partials);
    k_reduce_final  <<<3, 256, 0, stream>>>(partials, xbar);
    k_gate          <<<1, 32, 0, stream>>>(xbar, lora_route, task_id, meta_i, meta_g);
    k_build_weff    <<<(DIN * DOUT + 255) / 256, 256, 0, stream>>>(
                        W, lora_down, lora_up, meta_i, meta_g, weff_hi, weff_lo);

    dim3 grid(DOUT / 64, ROWS / 256);   // (12, 256); bx fastest -> L2 reuse of x strips
    k_gemm          <<<grid, 256, 0, stream>>>(x, weff_hi, weff_lo, out);
}
